// CustomLayerPCEN_51994874085772
// MI455X (gfx1250) — matrix operand order repaired and checked by reading, not compiled, not run on hardware
//
#include <hip/hip_runtime.h>
#include <hip/hip_bf16.h>
#include <stdint.h>

// PCEN for [F=1024, T=50000] fp32 spectrogram on gfx1250.
// EMA scan reformulated as 16x16x32 bf16 WMMA (coefficients are powers of two,
// exact in bf16; accumulation in f32; fp32 carry across 32-step tiles).
// Chunks made independent by a 64-sample warm-up (0.5^64 ~ 5e-20 << fp32 ulp).

typedef __attribute__((ext_vector_type(16))) __bf16 v16bf;
typedef __attribute__((ext_vector_type(8)))  float  v8f;

#define F_DIM   1024
#define T_DIM   50000
#define TCH     416                      // output timesteps per chunk
#define WARM    64                       // warm-up timesteps
#define WDATA   (WARM + TCH)             // 480 payload floats per LDS row
#define WIDTH   484                      // padded LDS row stride (floats)
#define ROWS    32                       // freq rows per block (2 waves x 16)
#define NCHUNK  ((T_DIM + TCH - 1) / TCH) // 121
#define EPS_F   1e-6f

// exact 2^e (e in [-126,127]) without transcendentals
__device__ __forceinline__ float pow2i(int e) {
    union { uint32_t u; float f; } c;
    c.u = (uint32_t)(e + 127) << 23;
    return c.f;
}
// native v_exp_f32 / v_log_f32 (both base-2 on AMDGPU)
__device__ __forceinline__ float fexp2(float x) { return __builtin_amdgcn_exp2f(x); }
__device__ __forceinline__ float flog2(float x) { return __builtin_amdgcn_logf(x); }

__global__ __launch_bounds__(64)
void pcen_wmma_kernel(const float* __restrict__ x,
                      const float* __restrict__ alphaP,
                      const float* __restrict__ rP,
                      const float* __restrict__ deltaP,
                      float* __restrict__ out)
{
    __shared__ float tile[ROWS * WIDTH];   // 61,952 B

    const int tid   = threadIdx.x;
    const int lane  = tid & 31;
    const int waveI = tid >> 5;            // 0..1
    const int chunk = blockIdx.x;          // 0..120
    const int fbase = blockIdx.y * ROWS;   // 0..992
    const int t0    = chunk * TCH;

    // ---- Stage 1: async global -> LDS staging (coalesced b128, ASYNCcnt) ----
    const uint32_t ldsBase = (uint32_t)(uintptr_t)(&tile[0]);
    #pragma unroll 4
    for (int it = 0; it < (ROWS * WDATA / 4) / 64; ++it) {   // 60 iterations
        int k   = tid + it * 64;
        int row = k / (WDATA / 4);
        int c4  = (k % (WDATA / 4)) * 4;
        int tg  = t0 - WARM + c4;
        tg = tg < 0 ? 0 : (tg > T_DIM - 4 ? T_DIM - 4 : tg);  // stays 16B aligned
        const float* gp = x + (size_t)(fbase + row) * T_DIM + tg;
        uint32_t lo = ldsBase + (uint32_t)(row * WIDTH + c4) * 4u;
        asm volatile("global_load_async_to_lds_b128 %0, %1, off"
                     :: "v"(lo), "v"((uint64_t)(uintptr_t)gp) : "memory");
    }
    asm volatile("s_wait_asynccnt 0x0" ::: "memory");

    if (chunk == 0) {
        // Zero the warm-up region: EMA starts from zero state.
        for (int i = 0; i < ROWS * WARM / 64; ++i) {
            int k = tid + i * 64;
            tile[(k / WARM) * WIDTH + (k % WARM)] = 0.0f;
        }
    }
    __syncthreads();

    // ---- Per-lane constants ----
    const int nn   = lane & 15;
    const int half = lane >> 4;
    const int b    = half * 8;

    // B = W^T scan coefficients: B[K=j][N=c] = W[i_out][j] = 0.5^(i_out-j+1), j<=i_out.
// Per-lane: N = nn, K = half*8 + e (e<8), 16 + half*8 + (e-8) (e>=8).  Exact in bf16 (powers of two).
    v16bf B1, B2;
    #pragma unroll
    for (int e = 0; e < 16; ++e) {
        int j = (e < 8) ? (half * 8 + e) : (16 + half * 8 + (e - 8));
        float w1 = (j <= nn)      ? pow2i(-(nn      - j + 1)) : 0.0f; // i_out = nn
        float w2 = (j <= nn + 16) ? pow2i(-(nn + 16 - j + 1)) : 0.0f; // i_out = nn+16
        B1[e] = (__bf16)w1;
        B2[e] = (__bf16)w2;
    }
    const float pA1 = pow2i(-(nn + 1));   // a^(i+1), i = nn      (tile 1)
    const float pA2 = pow2i(-(nn + 17));  // a^(i+1), i = 16+nn   (tile 2)

    const float nalpha = -alphaP[0];      // for x * 2^(-alpha*log2(eps+M))
    const float rr     = rP[0];
    const float delta  = deltaP[0];
    const float dr     = fexp2(rr * flog2(delta));   // delta^r

    float carry[8] = {0.f,0.f,0.f,0.f,0.f,0.f,0.f,0.f}; // fp32 EMA state per freq row

    const int rA = waveI * 16 + nn;   // A-matrix row (freq) this lane feeds

    // Per-lane output base: row (fbase + waveI*16 + half*8), column t0+nn.
    // Per-(v,blk) displacement v*T + blk*16 is a compile-time constant that
    // folds into the 24-bit store immediate; only tOff advances (by 32/step).
    float* const outBase =
        out + (size_t)(fbase + waveI * 16 + half * 8) * T_DIM + (t0 + nn);

    // ---- Stage 2: blocked scan via WMMA, then PCEN, over 15 steps of 32 t ----
    for (int s = 0; s < WDATA / 32; ++s) {
        // A = x tile, 16 freqs x 32 timesteps; lane holds K = [b,b+8) and [16+b,16+b+8)
        const float* rp = &tile[rA * WIDTH + s * 32];
        float4 x0 = *(const float4*)(rp + b);
        float4 x1 = *(const float4*)(rp + b + 4);
        float4 x2 = *(const float4*)(rp + b + 16);
        float4 x3 = *(const float4*)(rp + b + 20);
        v16bf A;
        A[0]=(__bf16)x0.x;  A[1]=(__bf16)x0.y;  A[2]=(__bf16)x0.z;  A[3]=(__bf16)x0.w;
        A[4]=(__bf16)x1.x;  A[5]=(__bf16)x1.y;  A[6]=(__bf16)x1.z;  A[7]=(__bf16)x1.w;
        A[8]=(__bf16)x2.x;  A[9]=(__bf16)x2.y;  A[10]=(__bf16)x2.z; A[11]=(__bf16)x2.w;
        A[12]=(__bf16)x3.x; A[13]=(__bf16)x3.y; A[14]=(__bf16)x3.z; A[15]=(__bf16)x3.w;

        v8f c0 = {};
        // D[f][i] = sum_j x[f][j] * W[i][j]   (f = VGPR row, i = lane column)
        v8f d1 = __builtin_amdgcn_wmma_f32_16x16x32_bf16(false, A, false, B1,
                                                         (short)0, c0, false, false);
        v8f d2 = __builtin_amdgcn_wmma_f32_16x16x32_bf16(false, A, false, B2,
                                                         (short)0, c0, false, false);

        float m1[8], m2[8];
        #pragma unroll
        for (int v = 0; v < 8; ++v) {
            m1[v] = fmaf(pA1, carry[v], d1[v]);       // + a^(i+1) * carry
            m2[v] = fmaf(pA2, carry[v], d2[v]);
            // next carry = M[f][31]: lane 15 holds f=v, lane 31 holds f=v+8
            carry[v] = __shfl(m2[v], (lane & 16) | 15, 32);
        }

        if (s >= WARM / 32) {                          // emit output (skip warm-up)
            const int tOff = (s - WARM / 32) * 32;     // uniform, +32 per step
            #pragma unroll
            for (int v = 0; v < 8; ++v) {
                const int rowL = waveI * 16 + half * 8 + v;  // LDS row of this reg
                // first 16 timesteps of the 32-step tile (d1)
                {
                    float xv  = tile[rowL * WIDTH + s * 32 + nn];
                    float inv = fexp2(nalpha * flog2(EPS_F + m1[v]));
                    float y   = fmaf(xv, inv, delta);
                    float o   = fexp2(rr * flog2(y)) - dr;
                    if (t0 + tOff + nn < T_DIM)
                        __builtin_nontemporal_store(o, outBase + (size_t)v * T_DIM + tOff);
                }
                // second 16 timesteps (d2)
                {
                    float xv  = tile[rowL * WIDTH + s * 32 + 16 + nn];
                    float inv = fexp2(nalpha * flog2(EPS_F + m2[v]));
                    float y   = fmaf(xv, inv, delta);
                    float o   = fexp2(rr * flog2(y)) - dr;
                    if (t0 + tOff + 16 + nn < T_DIM)
                        __builtin_nontemporal_store(o, outBase + (size_t)v * T_DIM + tOff + 16);
                }
            }
        }
    }
}

extern "C" void kernel_launch(void* const* d_in, const int* in_sizes, int n_in,
                              void* d_out, int out_size, void* d_ws, size_t ws_size,
                              hipStream_t stream) {
    (void)in_sizes; (void)n_in; (void)d_ws; (void)ws_size; (void)out_size;
    const float* data  = (const float*)d_in[0];
    const float* alpha = (const float*)d_in[1];
    const float* r     = (const float*)d_in[2];
    const float* delta = (const float*)d_in[3];
    float* out = (float*)d_out;

    dim3 grid(NCHUNK, F_DIM / ROWS);   // 121 x 32 blocks
    pcen_wmma_kernel<<<grid, 64, 0, stream>>>(data, alpha, r, delta, out);
}
